// OnnxSafeAttention_63453846831736
// MI455X (gfx1250) — matrix operand order repaired and checked by reading, not compiled, not run on hardware
//
#include <hip/hip_runtime.h>
#include <hip/hip_bf16.h>

// ---------------------------------------------------------------------------
// CDNA5 (gfx1250, wave32) fused multi-head attention.
//   B=4, S=2048, E=1024, H=16, D=64, SCALE=0.125
// Pipeline:
//   1) convert x, w_qkv, w_out  fp32 -> f16
//   2) QKV GEMM: fully-unrolled K loop, constant-offset b128 loads, 3-slot
//      distance-2 prefetch enforced with sched_group_barrier, wmma
//   3) flash attention per (b,h): QK^T wmma -> online softmax -> P*V wmma
//   4) output projection GEMM (same unrolled core), fp32 result
// ---------------------------------------------------------------------------

typedef __attribute__((ext_vector_type(16))) _Float16 v16h;
typedef __attribute__((ext_vector_type(8)))  float    v8f;
typedef __attribute__((ext_vector_type(4)))  _Float16 h4;

#define WMMA_F16(a, b, c) \
  __builtin_amdgcn_wmma_f32_16x16x32_f16(false, (a), false, (b), (short)0, (c), false, false)

// Scheduler group pins: 0x020 = VMEM read, 0x008 = MFMA/WMMA.
#if defined(__has_builtin)
#if __has_builtin(__builtin_amdgcn_sched_group_barrier)
#define SCHED_GROUP(mask, size, id) __builtin_amdgcn_sched_group_barrier((mask), (size), (id))
#endif
#endif
#ifndef SCHED_GROUP
#define SCHED_GROUP(mask, size, id)
#endif

// Contiguous 16-half fragment (B-operand layout run): two 16B loads at +0,+8.
__device__ __forceinline__ v16h load_frag_ofs(const _Float16* p, int e0) {
  v16h f;
  ((float4*)&f)[0] = *(const float4*)(p + e0);
  ((float4*)&f)[1] = *(const float4*)(p + e0 + 8);
  return f;
}

// A-operand fragment: contiguous 16-half run, two 16B loads at +0,+8.
__device__ __forceinline__ v16h load_afrag_ofs(const _Float16* p, int e0) {
  v16h f;
  ((float4*)&f)[0] = *(const float4*)(p + e0);
  ((float4*)&f)[1] = *(const float4*)(p + e0 + 8);
  return f;
}

// A-fragment with computed row/col (used for Q and LDS P staging).
// Lane l: row m=l&15, K = kbase..+15, kbase=(l>>4)*16.
__device__ __forceinline__ v16h load_a_frag(const _Float16* A, int lda,
                                            int m0, int k0) {
  const int lane = threadIdx.x & 31;
  const int m = m0 + (lane & 15);
  const int k = k0 + ((lane >> 4) << 4);
  return load_afrag_ofs(A + (size_t)m * lda + k, 0);
}

// ---------------------------------------------------------------------------
__global__ void __launch_bounds__(256) cvt_f32_f16_kernel(
    const float* __restrict__ in, _Float16* __restrict__ out, int n4) {
  int i = blockIdx.x * 256 + threadIdx.x;
  if (i < n4) {
    float4 v = ((const float4*)in)[i];
    h4 o;
    o[0] = (_Float16)v.x; o[1] = (_Float16)v.y;
    o[2] = (_Float16)v.z; o[3] = (_Float16)v.w;
    ((h4*)out)[i] = o;
  }
}

// ---------------------------------------------------------------------------
// GEMM core: 16x64 tile per wave over K=1024 (lda=ldb=1024), fully unrolled.
// 3-slot fragment file with distance-2 prefetch: iteration c consumes slot
// c%3 and refills slot (c+2)%3 (never read this or next iteration -> no WAR).
// sched_group_barrier pins the [10 VMEM reads][4 WMMA] alternation so the
// scheduler cannot re-serialize loads against their consuming WMMAs; ~20
// b128 loads remain in flight ahead of every WMMA group.
__device__ __forceinline__ void gemm_tile_16x64(
    const _Float16* __restrict__ A, const _Float16* __restrict__ Bt,
    int m0, int n0, v8f acc[4]) {
  const int lane = threadIdx.x & 31;
  const _Float16* pa = A + (size_t)(m0 + (lane & 15)) * 1024 + ((lane >> 4) << 4);
  const _Float16* pb = Bt + (size_t)(n0 + (lane & 15)) * 1024 + ((lane >> 4) << 4);

  v16h a[3], b0[3], b1[3], b2[3], b3[3];
#pragma unroll
  for (int s = 0; s < 2; ++s) {             // preload chunks 0,1
    const int o = s * 32;
    a[s]  = load_afrag_ofs(pa, o);
    b0[s] = load_frag_ofs(pb, o);
    b1[s] = load_frag_ofs(pb, 16384 + o);
    b2[s] = load_frag_ofs(pb, 32768 + o);
    b3[s] = load_frag_ofs(pb, 49152 + o);
  }

#pragma unroll
  for (int c = 0; c < 32; ++c) {            // 32 k-chunks of 32
    const int cur = c % 3;
    if (c + 2 < 32) {                       // prefetch chunk c+2
      const int nxt = (c + 2) % 3;
      const int o = (c + 2) * 32;
      a[nxt]  = load_afrag_ofs(pa, o);
      b0[nxt] = load_frag_ofs(pb, o);
      b1[nxt] = load_frag_ofs(pb, 16384 + o);
      b2[nxt] = load_frag_ofs(pb, 32768 + o);
      b3[nxt] = load_frag_ofs(pb, 49152 + o);
    }
    acc[0] = WMMA_F16(a[cur], b0[cur], acc[0]);
    acc[1] = WMMA_F16(a[cur], b1[cur], acc[1]);
    acc[2] = WMMA_F16(a[cur], b2[cur], acc[2]);
    acc[3] = WMMA_F16(a[cur], b3[cur], acc[3]);
    if (c + 2 < 32) SCHED_GROUP(0x020, 10, 0);  // 10 VMEM reads
    SCHED_GROUP(0x008, 4, 0);                   // 4 WMMA
  }
}

// ---------------------------------------------------------------------------
// QKV projection: qkv[m, f] = sum_e x[m,e] * w[f,e], M=8192, N=3072, K=1024.
// Output scattered into q[bh][s][d] (x 0.125), k[bh][s][d], vT[bh][d][s].
__global__ void __launch_bounds__(256, 1) qkv_gemm_kernel(
    const _Float16* __restrict__ x16, const _Float16* __restrict__ w16,
    _Float16* __restrict__ q, _Float16* __restrict__ k,
    _Float16* __restrict__ vT) {
  const int wid = (blockIdx.x * 256 + threadIdx.x) >> 5;   // 24576 waves
  const int mt = wid / 48;
  const int ng = wid - mt * 48;
  const int m0 = mt << 4;
  const int n0 = ng << 6;

  v8f acc[4] = {{}, {}, {}, {}};
  gemm_tile_16x64(x16, w16, m0, n0, acc);

  const int lane = threadIdx.x & 31;
  const int col = lane & 15;
  const int rbase = (lane >> 4) << 3;
#pragma unroll
  for (int t = 0; t < 4; ++t) {
    const int f = n0 + t * 16 + col;
    const int part = f >> 10;          // 0:q 1:k 2:v
    const int fi = f & 1023;
    const int hh = fi >> 6;
    const int dd = fi & 63;
#pragma unroll
    for (int r = 0; r < 8; ++r) {
      const int grow = m0 + rbase + r;      // global row in [0, 8192)
      const int bb = grow >> 11;
      const int ss = grow & 2047;
      const size_t bh = (size_t)(bb * 16 + hh);
      const float val = acc[t][r];
      if (part == 0)
        q[(bh * 2048 + ss) * 64 + dd] = (_Float16)(val * 0.125f);
      else if (part == 1)
        k[(bh * 2048 + ss) * 64 + dd] = (_Float16)val;
      else
        vT[(bh * 64 + dd) * 2048 + ss] = (_Float16)val;
    }
  }
}

// ---------------------------------------------------------------------------
// Flash attention. One block = 8 waves = one (b,h) x 128 query rows.
// Each wave owns a 16-row Q tile; loops over keys in tiles of 32. All K/V
// fragment loads use two base pointers (bumped once per tile) + immediate
// offsets. Next tile's K fragments refill in place during the softmax.
__global__ void __launch_bounds__(256) attn_kernel(
    const _Float16* __restrict__ q, const _Float16* __restrict__ k,
    const _Float16* __restrict__ vT, const unsigned char* __restrict__ mask,
    _Float16* __restrict__ attn) {
  __shared__ __align__(16) _Float16 pbuf[8][16 * 32];  // per-wave P staging

  const int wv = threadIdx.x >> 5;
  const int lane = threadIdx.x & 31;
  const int col = lane & 15;
  const int rbase = (lane >> 4) << 3;

  const int qblk = blockIdx.x & 15;
  const int bh = blockIdx.x >> 4;     // 0..63
  const int bb = bh >> 4;
  const int hh = bh & 15;
  const int q0 = qblk * 128 + wv * 16;

  const _Float16* Qh = q + (size_t)bh * 2048 * 64;   // [s][d], pre-scaled
  const _Float16* Kh = k + (size_t)bh * 2048 * 64;   // [s][d]
  const _Float16* Vt = vT + (size_t)bh * 64 * 2048;  // [d][s]
  const unsigned char* mb = mask + bb * 2048;

  const v16h qa0 = load_a_frag(Qh, 64, q0, 0);
  const v16h qa1 = load_a_frag(Qh, 64, q0, 32);

  v8f o0 = {}, o1 = {}, o2 = {}, o3 = {};
  float mrow[8], lrow[8];
#pragma unroll
  for (int r = 0; r < 8; ++r) { mrow[r] = -3.0e38f; lrow[r] = 0.0f; }

  _Float16* pw = &pbuf[wv][0];

  // per-lane fragment base pointers (B-operand layout)
  const _Float16* pK = Kh + (size_t)col * 64 + ((lane >> 4) << 4);
  const _Float16* pV = Vt + (size_t)col * 2048 + ((lane >> 4) << 4);

  // preload K fragments for the first key tile
  // offsets: second tile col = +16 keys * 64 = 1024; k0=32 within row
  v16h kf0 = load_frag_ofs(pK, 0);
  v16h kf1 = load_frag_ofs(pK, 32);
  v16h kf2 = load_frag_ofs(pK, 1024);
  v16h kf3 = load_frag_ofs(pK, 1056);

  for (int kb = 0; kb < 2048; kb += 32) {
    // V fragments for this tile (independent of scores): d-tiles 32768 apart
    v16h vb0 = load_frag_ofs(pV, 0);
    v16h vb1 = load_frag_ofs(pV, 32768);
    v16h vb2 = load_frag_ofs(pV, 65536);
    v16h vb3 = load_frag_ofs(pV, 98304);
    pV += 32;

    // scores: two 16x16 tiles over 32 keys, K-dim = D = 64 (2 wmma each)
    v8f s0 = {}, s1 = {};
    s0 = WMMA_F16(qa0, kf0, s0);
    s0 = WMMA_F16(qa1, kf1, s0);
    s1 = WMMA_F16(qa0, kf2, s1);
    s1 = WMMA_F16(qa1, kf3, s1);

    // refill K fragments for the next tile; they complete during softmax
    pK += 32 * 64;
    if (kb + 32 < 2048) {
      kf0 = load_frag_ofs(pK, 0);
      kf1 = load_frag_ofs(pK, 32);
      kf2 = load_frag_ofs(pK, 1024);
      kf3 = load_frag_ofs(pK, 1056);
    }

    const float madd0 = mb[kb + col] ? -3.0e38f : 0.0f;
    const float madd1 = mb[kb + 16 + col] ? -3.0e38f : 0.0f;

#pragma unroll
    for (int r = 0; r < 8; ++r) {
      const float e0 = s0[r] + madd0;
      const float e1 = s1[r] + madd1;
      // row max across the 16 lanes that hold this row (wave32 half-groups)
      float v = fmaxf(e0, e1);
      v = fmaxf(v, __shfl_xor(v, 1, 32));
      v = fmaxf(v, __shfl_xor(v, 2, 32));
      v = fmaxf(v, __shfl_xor(v, 4, 32));
      v = fmaxf(v, __shfl_xor(v, 8, 32));
      const float newm = fmaxf(mrow[r], v);
      const float corr = __expf(mrow[r] - newm);
      mrow[r] = newm;
      o0[r] *= corr; o1[r] *= corr; o2[r] *= corr; o3[r] *= corr;
      const float p0 = __expf(e0 - newm);
      const float p1 = __expf(e1 - newm);
      pw[(rbase + r) * 32 + col] = (_Float16)p0;
      pw[(rbase + r) * 32 + 16 + col] = (_Float16)p1;
      float ps = p0 + p1;
      ps += __shfl_xor(ps, 1, 32);
      ps += __shfl_xor(ps, 2, 32);
      ps += __shfl_xor(ps, 4, 32);
      ps += __shfl_xor(ps, 8, 32);
      lrow[r] = lrow[r] * corr + ps;
    }

    // P (16x32, A layout from LDS) x V (32 keys x 64 d, from vT columns)
    v16h pa = load_a_frag(pw, 32, 0, 0);
    o0 = WMMA_F16(pa, vb0, o0);
    o1 = WMMA_F16(pa, vb1, o1);
    o2 = WMMA_F16(pa, vb2, o2);
    o3 = WMMA_F16(pa, vb3, o3);
  }

  // normalize and store attn[b, s, h*64+d] as f16 for the output GEMM
#pragma unroll
  for (int r = 0; r < 8; ++r) {
    const float inv = 1.0f / lrow[r];
    const int ss = q0 + rbase + r;
    _Float16* dst = attn + ((size_t)(bb * 2048 + ss)) * 1024 + hh * 64;
    dst[0 * 16 + col] = (_Float16)(o0[r] * inv);
    dst[1 * 16 + col] = (_Float16)(o1[r] * inv);
    dst[2 * 16 + col] = (_Float16)(o2[r] * inv);
    dst[3 * 16 + col] = (_Float16)(o3[r] * inv);
  }
}

// ---------------------------------------------------------------------------
// Output projection: out[m, f] = sum_e attn[m,e] * w_out[f,e]; fp32 output.
__global__ void __launch_bounds__(256, 1) out_gemm_kernel(
    const _Float16* __restrict__ a16, const _Float16* __restrict__ w16,
    float* __restrict__ out) {
  const int wid = (blockIdx.x * 256 + threadIdx.x) >> 5;  // 8192 waves
  const int mt = wid >> 4;
  const int ng = wid & 15;
  const int m0 = mt << 4;
  const int n0 = ng << 6;

  v8f acc[4] = {{}, {}, {}, {}};
  gemm_tile_16x64(a16, w16, m0, n0, acc);

  const int lane = threadIdx.x & 31;
  const int col = lane & 15;
  const int rbase = (lane >> 4) << 3;
#pragma unroll
  for (int t = 0; t < 4; ++t)
#pragma unroll
    for (int r = 0; r < 8; ++r)
      out[(size_t)(m0 + rbase + r) * 1024 + n0 + t * 16 + col] = acc[t][r];
}

// ---------------------------------------------------------------------------
extern "C" void kernel_launch(void* const* d_in, const int* in_sizes, int n_in,
                              void* d_out, int out_size, void* d_ws, size_t ws_size,
                              hipStream_t stream) {
  (void)in_sizes; (void)n_in; (void)out_size; (void)ws_size;
  const float* x = (const float*)d_in[0];
  const unsigned char* mask = (const unsigned char*)d_in[1];  // bool (1 byte)
  const float* wqkv = (const float*)d_in[2];
  const float* wout = (const float*)d_in[3];
  float* out = (float*)d_out;

  char* p = (char*)d_ws;
  _Float16* x16  = (_Float16*)p; p += (size_t)8192 * 1024 * 2;   // 16 MB
  _Float16* wq16 = (_Float16*)p; p += (size_t)3072 * 1024 * 2;   //  6 MB
  _Float16* wo16 = (_Float16*)p; p += (size_t)1024 * 1024 * 2;   //  2 MB
  _Float16* q16  = (_Float16*)p; p += (size_t)64 * 2048 * 64 * 2; // 16 MB
  _Float16* k16  = (_Float16*)p; p += (size_t)64 * 2048 * 64 * 2; // 16 MB
  _Float16* vT16 = (_Float16*)p; p += (size_t)64 * 64 * 2048 * 2; // 16 MB
  _Float16* at16 = (_Float16*)p; p += (size_t)8192 * 1024 * 2;    // 16 MB

  cvt_f32_f16_kernel<<<(8192 * 1024 / 4 + 255) / 256, 256, 0, stream>>>(x, x16, 8192 * 1024 / 4);
  cvt_f32_f16_kernel<<<(3072 * 1024 / 4 + 255) / 256, 256, 0, stream>>>(wqkv, wq16, 3072 * 1024 / 4);
  cvt_f32_f16_kernel<<<(1024 * 1024 / 4 + 255) / 256, 256, 0, stream>>>(wout, wo16, 1024 * 1024 / 4);

  // 24576 waves / 8 per block
  qkv_gemm_kernel<<<3072, 256, 0, stream>>>(x16, wq16, q16, k16, vT16);
  // 64 (b,h) x 16 q-blocks
  attn_kernel<<<1024, 256, 0, stream>>>(q16, k16, vT16, mask, at16);
  // 8192 waves / 8 per block
  out_gemm_kernel<<<1024, 256, 0, stream>>>(at16, wo16, out);
}
